// Discriminator_18107582120628
// MI455X (gfx1250) — compile-verified, matrix operand order checked by reading, not run on hardware
//
#include <hip/hip_runtime.h>
#include <hip/hip_bf16.h>

// ---------------------------------------------------------------------------
// PointNet++ SA-MSG layer for MI455X (gfx1250, wave32, WMMA).
// Pipeline: fps -> ball query (3 radii) -> per-branch grouped MLP via
// v_wmma_f32_16x16x32_f16 -> max-pool -> 160->1 projection.
// ---------------------------------------------------------------------------

#define BATCH   8
#define NPTS    8192
#define NPOINT  1024
#define C_OUT   160

typedef __attribute__((ext_vector_type(16))) _Float16 v16h;
typedef __attribute__((ext_vector_type(8)))  _Float16 v8h;
typedef __attribute__((ext_vector_type(8)))  float    v8f;

union V16U { v16h v; v8h h[2]; };

// ---------------------------------------------------------------------------
// Kernel 1: farthest point sampling. One 1024-thread block per batch.
// Per-thread: 8 points' coords + running min-dist kept in registers.
// Argmax: shfl_xor wave reduce, then LDS cross-wave reduce (wave32).
// ---------------------------------------------------------------------------
__global__ __launch_bounds__(1024) void fps_kernel(const float* __restrict__ xyz,
                                                   int* __restrict__ fps_idx,
                                                   float* __restrict__ new_xyz) {
    const int b = blockIdx.x;
    const int t = threadIdx.x;
    const float* base = xyz + (size_t)b * NPTS * 3;

    float px[8], py[8], pz[8], dd[8];
#pragma unroll
    for (int k = 0; k < 8; ++k) {
        int p = t + k * 1024;
        px[k] = base[p * 3 + 0];
        py[k] = base[p * 3 + 1];
        pz[k] = base[p * 3 + 2];
        dd[k] = 1.0e10f;
    }

    __shared__ float s_v[32];
    __shared__ int   s_i[32];
    __shared__ int   s_far;

    int far = 0;
    for (int it = 0; it < NPOINT; ++it) {
        // centroid (same address across all lanes -> broadcast load)
        float cx = base[far * 3 + 0];
        float cy = base[far * 3 + 1];
        float cz = base[far * 3 + 2];
        if (t == 0) {
            fps_idx[b * NPOINT + it] = far;
            float* nx = new_xyz + ((size_t)b * NPOINT + it) * 3;
            nx[0] = cx; nx[1] = cy; nx[2] = cz;
        }
        float bv = -1.0f; int bi = 0;
#pragma unroll
        for (int k = 0; k < 8; ++k) {
            float dx = px[k] - cx, dy = py[k] - cy, dz = pz[k] - cz;
            float d = dx * dx + dy * dy + dz * dz;
            dd[k] = fminf(dd[k], d);
            if (dd[k] > bv) { bv = dd[k]; bi = t + k * 1024; }
        }
        // wave32 argmax reduce (prefer lower index on ties)
#pragma unroll
        for (int off = 16; off > 0; off >>= 1) {
            float ov = __shfl_xor(bv, off, 32);
            int   oi = __shfl_xor(bi, off, 32);
            if (ov > bv || (ov == bv && oi < bi)) { bv = ov; bi = oi; }
        }
        if ((t & 31) == 0) { s_v[t >> 5] = bv; s_i[t >> 5] = bi; }
        __syncthreads();
        if (t < 32) {
            bv = s_v[t]; bi = s_i[t];
#pragma unroll
            for (int off = 16; off > 0; off >>= 1) {
                float ov = __shfl_xor(bv, off, 32);
                int   oi = __shfl_xor(bi, off, 32);
                if (ov > bv || (ov == bv && oi < bi)) { bv = ov; bi = oi; }
            }
            if (t == 0) s_far = bi;
        }
        __syncthreads();
        far = s_far;
    }
}

// ---------------------------------------------------------------------------
// Kernel 2: ball query for all 3 radii in one pass. One wave32 per group.
// ballot + prefix popcount appends indices in ascending-index order,
// exactly matching the reference's top_k(-score) selection.
// ---------------------------------------------------------------------------
__device__ __forceinline__ void bq_append(unsigned m, int& cnt, int& first,
                                          int* __restrict__ arr, int cap,
                                          int p, int lane) {
    if (cnt == 0 && m != 0u) first = p - lane + __builtin_ctz(m);
    if ((m >> lane) & 1u) {
        int pos = cnt + __popc(m & ((1u << lane) - 1u));
        if (pos < cap) arr[pos] = p;
    }
    cnt += __popc(m);
}

__global__ __launch_bounds__(256) void ballquery_kernel(const float* __restrict__ xyz,
                                                        const float* __restrict__ new_xyz,
                                                        int* __restrict__ idx0,
                                                        int* __restrict__ idx1,
                                                        int* __restrict__ idx2) {
    const int wave = threadIdx.x >> 5;
    const int lane = threadIdx.x & 31;
    const int G = blockIdx.x * 8 + wave;         // group id in [0, B*NPOINT)
    const int b = G >> 10;

    const float* c = new_xyz + (size_t)G * 3;
    const float cx = c[0], cy = c[1], cz = c[2];
    const float* base = xyz + (size_t)b * NPTS * 3;

    int* arr0 = idx0 + (size_t)G * 16;
    int* arr1 = idx1 + (size_t)G * 32;
    int* arr2 = idx2 + (size_t)G * 128;

    int cnt0 = 0, cnt1 = 0, cnt2 = 0;
    int f0 = 0, f1 = 0, f2 = 0;

    for (int chunk = 0; chunk < NPTS; chunk += 32) {
        int p = chunk + lane;
        float dx = base[p * 3 + 0] - cx;
        float dy = base[p * 3 + 1] - cy;
        float dz = base[p * 3 + 2] - cz;
        float d2 = dx * dx + dy * dy + dz * dz;
        unsigned m2 = __builtin_amdgcn_ballot_w32(d2 < 0.16f);   // r=0.4
        unsigned m1 = __builtin_amdgcn_ballot_w32(d2 < 0.04f);   // r=0.2
        unsigned m0 = __builtin_amdgcn_ballot_w32(d2 < 0.01f);   // r=0.1
        bq_append(m0, cnt0, f0, arr0, 16,  p, lane);
        bq_append(m1, cnt1, f1, arr1, 32,  p, lane);
        bq_append(m2, cnt2, f2, arr2, 128, p, lane);
        if (cnt0 >= 16 && cnt1 >= 32 && cnt2 >= 128) break;  // uniform
    }
    // pad with first in-radius index (center itself always qualifies)
    for (int j = min(cnt0, 16) + lane; j < 16; j += 32) arr0[j] = f0;
    for (int j = min(cnt1, 32) + lane; j < 32; j += 32) arr1[j] = f1;
    for (int j = min(cnt2, 128) + lane; j < 128; j += 32) arr2[j] = f2;
}

// ---------------------------------------------------------------------------
// WMMA MLP layer. Per wave: one 16-row M tile, all output channels.
// actIn/actOut: [128][64] f16 LDS (zero-padded K). wT: [CO][KP] f16 LDS
// (transposed so B fragments load as two contiguous 16B ds reads, same as A).
// CDNA5 16-bit fragment layout: row = lane&15; lane&16 selects K half:
//   elements 0..7  -> K = kb..kb+7
//   elements 8..15 -> K = kb+16..kb+23,  kb = (lane&16) ? 8 : 0.
// ---------------------------------------------------------------------------
template <int KP, int CO>
__device__ __forceinline__ void mlp_layer(const _Float16* __restrict__ actIn,
                                          _Float16* __restrict__ actOut,
                                          const _Float16* __restrict__ wT,
                                          const float* __restrict__ bias,
                                          int rowBase, int lane) {
    const int rl = lane & 15;
    const int kb = (lane & 16) ? 8 : 0;
    const int ro = (lane & 16) ? 8 : 0;

    V16U a[KP / 32];
#pragma unroll
    for (int ks = 0; ks < KP / 32; ++ks) {
        const _Float16* pA = actIn + (rowBase + rl) * 64 + ks * 32 + kb;
        a[ks].h[0] = *(const v8h*)(pA);
        a[ks].h[1] = *(const v8h*)(pA + 16);
    }
#pragma unroll
    for (int nt = 0; nt < CO / 16; ++nt) {
        v8f acc = {};
#pragma unroll
        for (int ks = 0; ks < KP / 32; ++ks) {
            const _Float16* pB = wT + (nt * 16 + rl) * KP + ks * 32 + kb;
            V16U bf;
            bf.h[0] = *(const v8h*)(pB);
            bf.h[1] = *(const v8h*)(pB + 16);
            acc = __builtin_amdgcn_wmma_f32_16x16x32_f16(
                false, a[ks].v, false, bf.v, (short)0, acc, false, false);
        }
        const int n = nt * 16 + rl;
        const float bn = bias[n];
#pragma unroll
        for (int r = 0; r < 8; ++r) {
            float v = acc[r] + bn;
            v = fmaxf(v, 0.0f);
            actOut[(rowBase + ro + r) * 64 + n] = (_Float16)v;
        }
    }
}

// ---------------------------------------------------------------------------
// Kernel 3 (x3 branches): gather grouped points, 3-layer MLP, max-pool.
// 256 threads = 8 waves; GPB = 128/NS groups per iteration (128 rows).
// GITER iterations per block amortize the LDS weight/zero init.
// Loop-safety: the gather rewrites full zero-padded 32-col rows of bufA,
// and for every branch with GITER>1 each layer's K window is either fully
// rewritten each iteration or touched only by the one-time zero init.
// ---------------------------------------------------------------------------
template <int NS, int C1, int C2, int C3, int COFF, int GITER>
__global__ __launch_bounds__(256) void mlp_branch_kernel(
    const float* __restrict__ xyz, const float* __restrict__ new_xyz,
    const int* __restrict__ idx,
    const float* __restrict__ w0, const float* __restrict__ b0,
    const float* __restrict__ w1, const float* __restrict__ b1,
    const float* __restrict__ w2, const float* __restrict__ b2,
    float* __restrict__ feats) {
    constexpr int GPB = 128 / NS;
    constexpr int KP0 = 32;
    constexpr int KP1 = 32;                 // C1 <= 32
    constexpr int KP2 = (C2 > 32) ? 64 : 32;

    __shared__ __align__(16) _Float16 bufA[128 * 64];
    __shared__ __align__(16) _Float16 bufB[128 * 64];
    __shared__ __align__(16) _Float16 wT0[C1 * KP0];
    __shared__ __align__(16) _Float16 wT1[C2 * KP1];
    __shared__ __align__(16) _Float16 wT2[C3 * KP2];
    __shared__ float bias0[C1], bias1[C2], bias2[C3];

    const int t = threadIdx.x;

    // one-time init: zero activation buffers (padded K columns must be 0)
    {
        int* za = (int*)bufA;
        int* zb = (int*)bufB;
        for (int i = t; i < 128 * 64 / 2; i += 256) { za[i] = 0; zb[i] = 0; }
    }
    // transposed weights: wT[n*KP + k] = w[k*CO + n] (zero-padded K)
    for (int i = t; i < C1 * KP0; i += 256) {
        int n = i / KP0, k = i % KP0;
        wT0[i] = (k < 3) ? (_Float16)w0[k * C1 + n] : (_Float16)0.0f;
    }
    for (int i = t; i < C2 * KP1; i += 256) {
        int n = i / KP1, k = i % KP1;
        wT1[i] = (k < C1) ? (_Float16)w1[k * C2 + n] : (_Float16)0.0f;
    }
    for (int i = t; i < C3 * KP2; i += 256) {
        int n = i / KP2, k = i % KP2;
        wT2[i] = (k < C2) ? (_Float16)w2[k * C3 + n] : (_Float16)0.0f;
    }
    for (int i = t; i < C1; i += 256) bias0[i] = b0[i];
    for (int i = t; i < C2; i += 256) bias1[i] = b1[i];
    for (int i = t; i < C3; i += 256) bias2[i] = b2[i];
    __syncthreads();

    const int lane = t & 31;
    const int rowBase = (t >> 5) * 16;

    for (int g = 0; g < GITER; ++g) {
        const int Gbase = (blockIdx.x * GITER + g) * GPB;

        // gather grouped (relative) coords into bufA rows; write the full
        // zero-padded 32-col K window so iterations never see stale data.
        if (t < 128) {
            int gi = t / NS, j = t % NS;
            int G = Gbase + gi;
            int b = G >> 10;
            int pj = idx[(size_t)G * NS + j];
            const float* c = new_xyz + (size_t)G * 3;
            const float* p = xyz + ((size_t)b * NPTS + pj) * 3;
            union { _Float16 h[2]; int i; } u01, u23;
            u01.h[0] = (_Float16)(p[0] - c[0]);
            u01.h[1] = (_Float16)(p[1] - c[1]);
            u23.h[0] = (_Float16)(p[2] - c[2]);
            u23.h[1] = (_Float16)0.0f;
            int* r32 = (int*)(bufA + t * 64);
            r32[0] = u01.i;
            r32[1] = u23.i;
#pragma unroll
            for (int k = 2; k < 16; ++k) r32[k] = 0;
        }
        __syncthreads();

        mlp_layer<KP0, C1>(bufA, bufB, wT0, bias0, rowBase, lane);
        __syncthreads();
        mlp_layer<KP1, C2>(bufB, bufA, wT1, bias1, rowBase, lane);
        __syncthreads();
        mlp_layer<KP2, C3>(bufA, bufB, wT2, bias2, rowBase, lane);
        __syncthreads();

        // max-pool over the NS points of each group
        if (t < GPB * C3) {
            int gi = t / C3, ch = t % C3;
            int G = Gbase + gi;
            float m = 0.0f;  // post-ReLU values are >= 0
            const _Float16* col = bufB + gi * NS * 64 + ch;
#pragma unroll 4
            for (int j = 0; j < NS; ++j) m = fmaxf(m, (float)col[j * 64]);
            feats[(size_t)G * C_OUT + COFF + ch] = m;
        }
        __syncthreads();   // bufB consumed before next iteration's layer0
    }
}

// ---------------------------------------------------------------------------
// Kernel 4: out[b,0,s] = dot(feats[b,s,:160], w_out) + b_out
// ---------------------------------------------------------------------------
__global__ __launch_bounds__(256) void final_proj_kernel(const float* __restrict__ feats,
                                                         const float* __restrict__ w_out,
                                                         const float* __restrict__ b_out,
                                                         float* __restrict__ out) {
    int t = blockIdx.x * blockDim.x + threadIdx.x;
    if (t >= BATCH * NPOINT) return;
    const float* f = feats + (size_t)t * C_OUT;
    float s = b_out[0];
#pragma unroll 8
    for (int c = 0; c < C_OUT; ++c) s += f[c] * w_out[c];
    out[t] = s;
}

// ---------------------------------------------------------------------------
extern "C" void kernel_launch(void* const* d_in, const int* in_sizes, int n_in,
                              void* d_out, int out_size, void* d_ws, size_t ws_size,
                              hipStream_t stream) {
    const float* xyz  = (const float*)d_in[0];
    const float* w0_0 = (const float*)d_in[1];  const float* b0_0 = (const float*)d_in[2];
    const float* w0_1 = (const float*)d_in[3];  const float* b0_1 = (const float*)d_in[4];
    const float* w0_2 = (const float*)d_in[5];  const float* b0_2 = (const float*)d_in[6];
    const float* w1_0 = (const float*)d_in[7];  const float* b1_0 = (const float*)d_in[8];
    const float* w1_1 = (const float*)d_in[9];  const float* b1_1 = (const float*)d_in[10];
    const float* w1_2 = (const float*)d_in[11]; const float* b1_2 = (const float*)d_in[12];
    const float* w2_0 = (const float*)d_in[13]; const float* b2_0 = (const float*)d_in[14];
    const float* w2_1 = (const float*)d_in[15]; const float* b2_1 = (const float*)d_in[16];
    const float* w2_2 = (const float*)d_in[17]; const float* b2_2 = (const float*)d_in[18];
    const float* w_out = (const float*)d_in[19];
    const float* b_out = (const float*)d_in[20];
    float* out = (float*)d_out;

    // workspace layout
    char* ws = (char*)d_ws;
    int*   fps_idx = (int*)ws;                         ws += (size_t)BATCH * NPOINT * sizeof(int);
    float* new_xyz = (float*)ws;                       ws += (size_t)BATCH * NPOINT * 3 * sizeof(float);
    int*   idx0    = (int*)ws;                         ws += (size_t)BATCH * NPOINT * 16 * sizeof(int);
    int*   idx1    = (int*)ws;                         ws += (size_t)BATCH * NPOINT * 32 * sizeof(int);
    int*   idx2    = (int*)ws;                         ws += (size_t)BATCH * NPOINT * 128 * sizeof(int);
    float* feats   = (float*)ws;                       ws += (size_t)BATCH * NPOINT * C_OUT * sizeof(float);

    fps_kernel<<<BATCH, 1024, 0, stream>>>(xyz, fps_idx, new_xyz);
    ballquery_kernel<<<(BATCH * NPOINT) / 8, 256, 0, stream>>>(xyz, new_xyz, idx0, idx1, idx2);

    // branch 0: nsample=16, 3->16->16->32, offset 0, 8 groups/iter, 1 iter
    mlp_branch_kernel<16, 16, 16, 32, 0, 1><<<(BATCH * NPOINT) / 8, 256, 0, stream>>>(
        xyz, new_xyz, idx0, w0_0, b0_0, w0_1, b0_1, w0_2, b0_2, feats);
    // branch 1: nsample=32, 3->32->32->64, offset 32, 4 groups/iter, 2 iters
    mlp_branch_kernel<32, 32, 32, 64, 32, 2><<<(BATCH * NPOINT) / 8, 256, 0, stream>>>(
        xyz, new_xyz, idx1, w1_0, b1_0, w1_1, b1_1, w1_2, b1_2, feats);
    // branch 2: nsample=128, 3->32->48->64, offset 96, 1 group/iter, 8 iters
    mlp_branch_kernel<128, 32, 48, 64, 96, 8><<<(BATCH * NPOINT) / 8, 256, 0, stream>>>(
        xyz, new_xyz, idx2, w2_0, b2_0, w2_1, b2_1, w2_2, b2_2, feats);

    final_proj_kernel<<<(BATCH * NPOINT + 255) / 256, 256, 0, stream>>>(feats, w_out, b_out, out);
}
